// JointSpatioTemporalBlock_73787538145515
// MI455X (gfx1250) — matrix operand order repaired and checked by reading, not compiled, not run on hardware
//
#include <hip/hip_runtime.h>
#include <hip/hip_bf16.h>
#include <math.h>

typedef __attribute__((ext_vector_type(16))) _Float16 v16h;
typedef __attribute__((ext_vector_type(8)))  _Float16 v8h;
typedef __attribute__((ext_vector_type(8)))  float    v8f;

#define BB    2
#define TT    100
#define VSZ   25
#define DD    256
#define HH    8
#define HDIM  32
#define LL    2500          // T*V
#define LPAD  2528          // multiple of 32 (79 key blocks, 79 q super-tiles)
#define DFF   1024
#define MAXT  300
#define RELN  (2*MAXT-1)    // 599
#define QK_SCALE 0.17677669529663687f   // 1/sqrt(32)
#define LN_EPS 1e-5f

// ---------------------------------------------------------------- helpers

__device__ __forceinline__ v16h pack16(v8h lo, v8h hi) {
  v16h r;
#pragma unroll
  for (int i = 0; i < 8; ++i) { r[i] = lo[i]; r[8 + i] = hi[i]; }
  return r;
}

__device__ __forceinline__ float redmax16(float v) {
  v = fmaxf(v, __shfl_xor(v, 1, 32));
  v = fmaxf(v, __shfl_xor(v, 2, 32));
  v = fmaxf(v, __shfl_xor(v, 4, 32));
  v = fmaxf(v, __shfl_xor(v, 8, 32));
  return v;
}
__device__ __forceinline__ float redsum16(float v) {
  v += __shfl_xor(v, 1, 32);
  v += __shfl_xor(v, 2, 32);
  v += __shfl_xor(v, 4, 32);
  v += __shfl_xor(v, 8, 32);
  return v;
}

// Fragment layouts per CDNA5 ISA 7.12.2 (wave32):
//   A: lane row = lane%16; halves hold K {8h..8h+7, 16+8h..16+8h+7}
//   B: lane col = lane%16; halves hold K {8h..8h+7, 16+8h..16+8h+7} (from Bt row)
__device__ __forceinline__ v16h load_afrag(const _Float16* arow, int k0, int hf) {
  return pack16(*(const v8h*)(arow + k0 + 8 * hf),
                *(const v8h*)(arow + k0 + 16 + 8 * hf));
}
__device__ __forceinline__ v16h load_bfrag(const _Float16* brow, int k0, int hf) {
  return pack16(*(const v8h*)(brow + k0 + 8 * hf),
                *(const v8h*)(brow + k0 + 16 + 8 * hf));
}

// Wave-level 32x32 f32 tile (2x2 register blocking -> 4 independent WMMAs per
// k-step: doubles L2 arithmetic intensity and hides XDL latency without NOPs).
// A: 32xK f16 row-major (lda).  Bt = B^T: 32 rows of K f16 (ldb).
__device__ __forceinline__ void wmma_tile32(const _Float16* __restrict__ A, int lda,
                                            const _Float16* __restrict__ Bt, int ldb,
                                            int K, int lane, v8f acc[2][2]) {
  const int hf = lane >> 4, ln = lane & 15;
  const _Float16* arow0 = A + (size_t)ln * lda;
  const _Float16* arow1 = A + (size_t)(16 + ln) * lda;
  const _Float16* brow0 = Bt + (size_t)ln * ldb;
  const _Float16* brow1 = Bt + (size_t)(16 + ln) * ldb;
  for (int k0 = 0; k0 < K; k0 += 32) {
    const v16h a0 = load_afrag(arow0, k0, hf);
    const v16h a1 = load_afrag(arow1, k0, hf);
    const v16h b0 = load_bfrag(brow0, k0, hf);
    const v16h b1 = load_bfrag(brow1, k0, hf);
    acc[0][0] = __builtin_amdgcn_wmma_f32_16x16x32_f16(false, a0, false, b0, (short)0, acc[0][0], false, false);
    acc[0][1] = __builtin_amdgcn_wmma_f32_16x16x32_f16(false, a0, false, b1, (short)0, acc[0][1], false, false);
    acc[1][0] = __builtin_amdgcn_wmma_f32_16x16x32_f16(false, a1, false, b0, (short)0, acc[1][0], false, false);
    acc[1][1] = __builtin_amdgcn_wmma_f32_16x16x32_f16(false, a1, false, b1, (short)0, acc[1][1], false, false);
  }
}

// ---------------------------------------------------------------- kernels

// f32 (K,N) -> f16 transposed (N,K)
__global__ void cvt_transpose(const float* __restrict__ in, _Float16* __restrict__ out,
                              int K, int N) {
  int i = blockIdx.x * blockDim.x + threadIdx.x;
  if (i >= K * N) return;
  int k = i / N, n = i - k * N;
  out[(size_t)n * K + k] = (_Float16)in[i];
}

// LayerNorm: one row per block (D=256 == blockDim).  Rows l>=LL zero-padded.
__global__ void __launch_bounds__(256) ln_kernel(const float* __restrict__ in,
                                                 const float* __restrict__ g,
                                                 const float* __restrict__ be,
                                                 _Float16* __restrict__ out,
                                                 int in_rows) {
  __shared__ float red[256];
  const int r = blockIdx.x;
  const int b = r / LPAD, l = r % LPAD;
  const int tid = threadIdx.x;
  const bool valid = l < LL;
  float v = 0.f;
  if (valid) v = in[((size_t)b * in_rows + l) * DD + tid];
  red[tid] = v;
  __syncthreads();
  for (int s = 128; s > 0; s >>= 1) { if (tid < s) red[tid] += red[tid + s]; __syncthreads(); }
  const float mean = red[0] * (1.0f / DD);
  __syncthreads();
  const float dv = v - mean;
  red[tid] = dv * dv;
  __syncthreads();
  for (int s = 128; s > 0; s >>= 1) { if (tid < s) red[tid] += red[tid + s]; __syncthreads(); }
  const float var = red[0] * (1.0f / DD);
  const float o = valid ? (dv * rsqrtf(var + LN_EPS) * g[tid] + be[tid]) : 0.f;
  out[(size_t)r * DD + tid] = (_Float16)o;
}

// h1 (B*LPAD x 256 f16) @ wqkv^T -> q (scaled), k, v^T in f16 per-head layouts
__global__ void __launch_bounds__(256) gemm_qkv(const _Float16* __restrict__ h1,
                                                const _Float16* __restrict__ wt,
                                                const float* __restrict__ bqkv,
                                                _Float16* __restrict__ qf,
                                                _Float16* __restrict__ kf,
                                                _Float16* __restrict__ vt) {
  const int NT = 768 / 32;             // 24
  const int MT = (BB * LPAD) / 32;     // 158
  const int wave = threadIdx.x >> 5, lane = threadIdx.x & 31;
  const int t = blockIdx.x * 8 + wave;
  if (t >= MT * NT) return;
  const int mt = t / NT, nt = t % NT;
  const int m0 = mt * 32, c0 = nt * 32;
  v8f acc[2][2] = {};
  wmma_tile32(h1 + (size_t)m0 * DD, DD, wt + (size_t)c0 * DD, DD, DD, lane, acc);
  const int hf = lane >> 4, ln = lane & 15;
  const int b = m0 / LPAD;
#pragma unroll
  for (int tj = 0; tj < 2; ++tj) {
    const int c = c0 + 16 * tj + ln;
    const int sec = c >> 8;            // 0=q 1=k 2=v
    const int head = (c & 255) >> 5;
    const int hd = c & 31;
    const float bias = bqkv[c];
#pragma unroll
    for (int ti = 0; ti < 2; ++ti) {
#pragma unroll
      for (int j = 0; j < 8; ++j) {
        const int r = m0 + 16 * ti + j + 8 * hf;
        const int l = r - b * LPAD;
        const float v = acc[ti][tj][j] + bias;
        if (sec == 0)
          qf[((size_t)(b * HH + head) * LPAD + l) * HDIM + hd] = (_Float16)(v * QK_SCALE);
        else if (sec == 1)
          kf[((size_t)(b * HH + head) * LPAD + l) * HDIM + hd] = (_Float16)v;
        else
          vt[((size_t)(b * HH + head) * HDIM + hd) * LPAD + l] = (_Float16)v;
      }
    }
  }
}

// Online-softmax bookkeeping for one 16x32 score tile; writes P (f16) to LDS.
__device__ __forceinline__ void fa_update(const v8f& s0, const v8f& s1,
                                          bool maskA, bool maskB,
                                          int ktA, int kvA, int ktB, int kvB,
                                          const int tq[8], const int vq[8],
                                          const float* rel_s, const float* gb_s,
                                          float mr[8], float lr[8],
                                          v8f& o0, v8f& o1,
                                          _Float16* myp, int hf, int ln) {
#pragma unroll
  for (int j = 0; j < 8; ++j) {
    float sa = maskA ? -1e30f
                     : (s0[j] + rel_s[tq[j] - ktA + (MAXT - 1)] + gb_s[vq[j] * VSZ + kvA]);
    float sb = maskB ? -1e30f
                     : (s1[j] + rel_s[tq[j] - ktB + (MAXT - 1)] + gb_s[vq[j] * VSZ + kvB]);
    const float mnew = fmaxf(mr[j], redmax16(fmaxf(sa, sb)));
    const float sc = __expf(mr[j] - mnew);
    const float pa = __expf(sa - mnew);
    const float pb = __expf(sb - mnew);
    lr[j] = lr[j] * sc + redsum16(pa + pb);
    o0[j] *= sc;
    o1[j] *= sc;
    mr[j] = mnew;
    myp[(j + 8 * hf) * 32 + ln]      = (_Float16)pa;
    myp[(j + 8 * hf) * 32 + 16 + ln] = (_Float16)pb;
  }
}

// Flash attention: one wave per 32-query super-tile (2 q-tiles share K/V
// fragments -> half the L2 K/V traffic, 8 WMMAs per 32-key block).
__global__ void __launch_bounds__(128) attn_kernel(const _Float16* __restrict__ qf,
                                                   const _Float16* __restrict__ kf,
                                                   const _Float16* __restrict__ vt,
                                                   const float* __restrict__ relp,
                                                   const float* __restrict__ gbias,
                                                   _Float16* __restrict__ attn_h) {
  constexpr int WV = 4;
  constexpr int QT = LPAD / 32;            // 79
  constexpr int QB = (QT + WV - 1) / WV;   // 20
  __shared__ float rel_s[RELN];
  __shared__ float gb_s[VSZ * VSZ];
  __shared__ _Float16 pt[WV][2][16 * 32];

  const int qb = blockIdx.x % QB;
  const int h  = (blockIdx.x / QB) % HH;
  const int b  = blockIdx.x / (QB * HH);

  for (int i = threadIdx.x; i < RELN; i += blockDim.x)      rel_s[i] = relp[h * RELN + i];
  for (int i = threadIdx.x; i < VSZ * VSZ; i += blockDim.x) gb_s[i]  = gbias[h * VSZ * VSZ + i];
  __syncthreads();

  const int wave = threadIdx.x >> 5;
  const int lane = threadIdx.x & 31;
  const int qt = qb * WV + wave;
  if (qt >= QT) return;
  const int q0 = qt * 32;
  const int hf = lane >> 4, ln = lane & 15;

  int tqA[8], vqA[8], tqB[8], vqB[8];
#pragma unroll
  for (int j = 0; j < 8; ++j) {
    int la = q0 + j + 8 * hf;       if (la > LL - 1) la = LL - 1;
    int lb = q0 + 16 + j + 8 * hf;  if (lb > LL - 1) lb = LL - 1;
    tqA[j] = la / VSZ; vqA[j] = la - tqA[j] * VSZ;
    tqB[j] = lb / VSZ; vqB[j] = lb - tqB[j] * VSZ;
  }

  const _Float16* qbase = qf + (size_t)(b * HH + h) * LPAD * HDIM;
  const v16h aqA = load_afrag(qbase + (size_t)(q0 + ln) * HDIM, 0, hf);
  const v16h aqB = load_afrag(qbase + (size_t)(q0 + 16 + ln) * HDIM, 0, hf);

  const _Float16* kbase = kf + (size_t)(b * HH + h) * LPAD * HDIM;
  const _Float16* vbase = vt + (size_t)(b * HH + h) * HDIM * LPAD;
  _Float16* mypA = pt[wave][0];
  _Float16* mypB = pt[wave][1];

  float mrA[8], lrA[8], mrB[8], lrB[8];
#pragma unroll
  for (int j = 0; j < 8; ++j) { mrA[j] = mrB[j] = -1e30f; lrA[j] = lrB[j] = 0.f; }
  v8f o0A = {}, o1A = {}, o0B = {}, o1B = {};

  for (int key0 = 0; key0 < LPAD; key0 += 32) {
    // shared K fragments (B-operands of Q@K^T), K-dim = HD = 32
    const v16h bk0 = load_bfrag(kbase + (size_t)(key0 + ln) * HDIM, 0, hf);
    const v16h bk1 = load_bfrag(kbase + (size_t)(key0 + 16 + ln) * HDIM, 0, hf);
    const v8f zero = {};
    v8f s0A = __builtin_amdgcn_wmma_f32_16x16x32_f16(false, aqA, false, bk0, (short)0, zero, false, false);
    v8f s1A = __builtin_amdgcn_wmma_f32_16x16x32_f16(false, aqA, false, bk1, (short)0, zero, false, false);
    v8f s0B = __builtin_amdgcn_wmma_f32_16x16x32_f16(false, aqB, false, bk0, (short)0, zero, false, false);
    v8f s1B = __builtin_amdgcn_wmma_f32_16x16x32_f16(false, aqB, false, bk1, (short)0, zero, false, false);

    const int ka = key0 + ln;
    const int kb = key0 + 16 + ln;
    const bool maskA = ka >= LL, maskB = kb >= LL;
    const int kaC = maskA ? LL - 1 : ka;
    const int kbC = maskB ? LL - 1 : kb;
    const int ktA = kaC / VSZ, kvA = kaC - ktA * VSZ;
    const int ktB = kbC / VSZ, kvB = kbC - ktB * VSZ;

    fa_update(s0A, s1A, maskA, maskB, ktA, kvA, ktB, kvB, tqA, vqA,
              rel_s, gb_s, mrA, lrA, o0A, o1A, mypA, hf, ln);
    fa_update(s0B, s1B, maskA, maskB, ktA, kvA, ktB, kvB, tqB, vqB,
              rel_s, gb_s, mrB, lrB, o0B, o1B, mypB, hf, ln);
    __builtin_amdgcn_wave_barrier();
    // re-read P in A-fragment layout (LDS in-order within a wave)
    const v16h apA = load_afrag(mypA + ln * 32, 0, hf);
    const v16h apB = load_afrag(mypB + ln * 32, 0, hf);
    __builtin_amdgcn_wave_barrier();
    // shared V fragments (V stored transposed HD x LPAD -> contiguous loads)
    const v16h bv0 = load_bfrag(vbase + (size_t)ln * LPAD + key0, 0, hf);
    const v16h bv1 = load_bfrag(vbase + (size_t)(16 + ln) * LPAD + key0, 0, hf);
    o0A = __builtin_amdgcn_wmma_f32_16x16x32_f16(false, apA, false, bv0, (short)0, o0A, false, false);
    o1A = __builtin_amdgcn_wmma_f32_16x16x32_f16(false, apA, false, bv1, (short)0, o1A, false, false);
    o0B = __builtin_amdgcn_wmma_f32_16x16x32_f16(false, apB, false, bv0, (short)0, o0B, false, false);
    o1B = __builtin_amdgcn_wmma_f32_16x16x32_f16(false, apB, false, bv1, (short)0, o1B, false, false);
  }

#pragma unroll
  for (int j = 0; j < 8; ++j) {
    const int la = q0 + j + 8 * hf;
    const int lb = q0 + 16 + j + 8 * hf;
    const float invA = 1.0f / lrA[j];
    const float invB = 1.0f / lrB[j];
    _Float16* dstA = attn_h + (size_t)(b * LPAD + la) * DD + h * HDIM;
    _Float16* dstB = attn_h + (size_t)(b * LPAD + lb) * DD + h * HDIM;
    dstA[ln]      = (_Float16)(o0A[j] * invA);
    dstA[16 + ln] = (_Float16)(o1A[j] * invA);
    dstB[ln]      = (_Float16)(o0B[j] * invB);
    dstB[16 + ln] = (_Float16)(o1B[j] * invB);
  }
}

// attn_out @ w_proj + b_proj + residual(x) -> x1 (f32, padded rows zeroed)
__global__ void __launch_bounds__(256) gemm_proj(const _Float16* __restrict__ ah,
                                                 const _Float16* __restrict__ wt,
                                                 const float* __restrict__ bp,
                                                 const float* __restrict__ xin,
                                                 float* __restrict__ x1) {
  const int NT = DD / 32;              // 8
  const int MT = (BB * LPAD) / 32;     // 158
  const int wave = threadIdx.x >> 5, lane = threadIdx.x & 31;
  const int t = blockIdx.x * 8 + wave;
  if (t >= MT * NT) return;
  const int mt = t / NT, nt = t % NT;
  const int m0 = mt * 32, c0 = nt * 32;
  v8f acc[2][2] = {};
  wmma_tile32(ah + (size_t)m0 * DD, DD, wt + (size_t)c0 * DD, DD, DD, lane, acc);
  const int hf = lane >> 4, ln = lane & 15;
  const int b = m0 / LPAD;
#pragma unroll
  for (int tj = 0; tj < 2; ++tj) {
    const int d = c0 + 16 * tj + ln;
    const float bias = bp[d];
#pragma unroll
    for (int ti = 0; ti < 2; ++ti) {
#pragma unroll
      for (int j = 0; j < 8; ++j) {
        const int r = m0 + 16 * ti + j + 8 * hf;
        const int l = r - b * LPAD;
        float v = 0.f;
        if (l < LL) v = xin[((size_t)b * LL + l) * DD + d] + acc[ti][tj][j] + bias;
        x1[(size_t)r * DD + d] = v;
      }
    }
  }
}

// h2 @ w1 + b1 -> exact GELU -> f16
__global__ void __launch_bounds__(256) gemm_ffn1(const _Float16* __restrict__ h2,
                                                 const _Float16* __restrict__ wt,
                                                 const float* __restrict__ bias1,
                                                 _Float16* __restrict__ ff) {
  const int NT = DFF / 32;             // 32
  const int MT = (BB * LPAD) / 32;
  const int wave = threadIdx.x >> 5, lane = threadIdx.x & 31;
  const int t = blockIdx.x * 8 + wave;
  if (t >= MT * NT) return;
  const int mt = t / NT, nt = t % NT;
  const int m0 = mt * 32, c0 = nt * 32;
  v8f acc[2][2] = {};
  wmma_tile32(h2 + (size_t)m0 * DD, DD, wt + (size_t)c0 * DD, DD, DD, lane, acc);
  const int hf = lane >> 4, ln = lane & 15;
#pragma unroll
  for (int tj = 0; tj < 2; ++tj) {
    const int c = c0 + 16 * tj + ln;
    const float bias = bias1[c];
#pragma unroll
    for (int ti = 0; ti < 2; ++ti) {
#pragma unroll
      for (int j = 0; j < 8; ++j) {
        const int r = m0 + 16 * ti + j + 8 * hf;
        const float v = acc[ti][tj][j] + bias;
        const float ge = 0.5f * v * (1.f + erff(v * 0.70710678118654752f));
        ff[(size_t)r * DFF + c] = (_Float16)ge;
      }
    }
  }
}

// ff @ w2 + b2 + x1 -> final output (only valid rows)
__global__ void __launch_bounds__(256) gemm_ffn2(const _Float16* __restrict__ ff,
                                                 const _Float16* __restrict__ wt,
                                                 const float* __restrict__ bias2,
                                                 const float* __restrict__ x1,
                                                 float* __restrict__ out) {
  const int NT = DD / 32;              // 8
  const int MT = (BB * LPAD) / 32;
  const int wave = threadIdx.x >> 5, lane = threadIdx.x & 31;
  const int t = blockIdx.x * 8 + wave;
  if (t >= MT * NT) return;
  const int mt = t / NT, nt = t % NT;
  const int m0 = mt * 32, c0 = nt * 32;
  v8f acc[2][2] = {};
  wmma_tile32(ff + (size_t)m0 * DFF, DFF, wt + (size_t)c0 * DFF, DFF, DFF, lane, acc);
  const int hf = lane >> 4, ln = lane & 15;
  const int b = m0 / LPAD;
#pragma unroll
  for (int tj = 0; tj < 2; ++tj) {
    const int d = c0 + 16 * tj + ln;
    const float bias = bias2[d];
#pragma unroll
    for (int ti = 0; ti < 2; ++ti) {
#pragma unroll
      for (int j = 0; j < 8; ++j) {
        const int r = m0 + 16 * ti + j + 8 * hf;
        const int l = r - b * LPAD;
        if (l < LL)
          out[((size_t)b * LL + l) * DD + d] = acc[ti][tj][j] + bias + x1[(size_t)r * DD + d];
      }
    }
  }
}

// ---------------------------------------------------------------- launch

extern "C" void kernel_launch(void* const* d_in, const int* in_sizes, int n_in,
                              void* d_out, int out_size, void* d_ws, size_t ws_size,
                              hipStream_t stream) {
  (void)in_sizes; (void)n_in; (void)out_size; (void)ws_size;
  const float* x      = (const float*)d_in[0];
  const float* ln1_g  = (const float*)d_in[1];
  const float* ln1_b  = (const float*)d_in[2];
  const float* w_qkv  = (const float*)d_in[3];
  const float* b_qkv  = (const float*)d_in[4];
  const float* relp   = (const float*)d_in[5];
  const float* gbias  = (const float*)d_in[6];
  const float* w_proj = (const float*)d_in[7];
  const float* b_proj = (const float*)d_in[8];
  const float* ln2_g  = (const float*)d_in[9];
  const float* ln2_b  = (const float*)d_in[10];
  const float* w1     = (const float*)d_in[11];
  const float* b1     = (const float*)d_in[12];
  const float* w2     = (const float*)d_in[13];
  const float* b2     = (const float*)d_in[14];
  float* out = (float*)d_out;

  char* wsp = (char*)d_ws;
  auto take = [&](size_t bytes) -> void* {
    void* r = (void*)wsp;
    wsp += (bytes + 255) & ~(size_t)255;
    return r;
  };
  const size_t ROWS = (size_t)BB * LPAD;                       // 5056
  _Float16* wqkv_t  = (_Float16*)take((size_t)768 * 256 * 2);  // (3D, D) transposed
  _Float16* wproj_t = (_Float16*)take((size_t)256 * 256 * 2);
  _Float16* w1_t    = (_Float16*)take((size_t)1024 * 256 * 2);
  _Float16* w2_t    = (_Float16*)take((size_t)256 * 1024 * 2);
  _Float16* h1_h    = (_Float16*)take(ROWS * DD * 2);
  _Float16* qf      = (_Float16*)take(ROWS * DD * 2);  // B,H,LPAD,HD
  _Float16* kf      = (_Float16*)take(ROWS * DD * 2);  // B,H,LPAD,HD
  _Float16* vtr     = (_Float16*)take(ROWS * DD * 2);  // B,H,HD,LPAD
  _Float16* attn_h  = (_Float16*)take(ROWS * DD * 2);
  float*    x1      = (float*)take(ROWS * DD * 4);
  _Float16* h2_h    = (_Float16*)take(ROWS * DD * 2);
  _Float16* ff1_h   = (_Float16*)take(ROWS * DFF * 2);

  cvt_transpose<<<(256 * 768  + 255) / 256, 256, 0, stream>>>(w_qkv,  wqkv_t,  256, 768);
  cvt_transpose<<<(256 * 256  + 255) / 256, 256, 0, stream>>>(w_proj, wproj_t, 256, 256);
  cvt_transpose<<<(256 * 1024 + 255) / 256, 256, 0, stream>>>(w1,     w1_t,    256, 1024);
  cvt_transpose<<<(1024 * 256 + 255) / 256, 256, 0, stream>>>(w2,     w2_t,    1024, 256);

  ln_kernel<<<BB * LPAD, 256, 0, stream>>>(x, ln1_g, ln1_b, h1_h, LL);

  const int MT = (BB * LPAD) / 32;  // 158
  gemm_qkv<<<(MT * 24 + 7) / 8, 256, 0, stream>>>(h1_h, wqkv_t, b_qkv, qf, kf, vtr);

  const int QB = ((LPAD / 32) + 3) / 4;  // 20 blocks of 4 super-tile waves
  attn_kernel<<<BB * HH * QB, 128, 0, stream>>>(qf, kf, vtr, relp, gbias, attn_h);

  gemm_proj<<<(MT * 8 + 7) / 8, 256, 0, stream>>>(attn_h, wproj_t, b_proj, x, x1);
  ln_kernel<<<BB * LPAD, 256, 0, stream>>>(x1, ln2_g, ln2_b, h2_h, LPAD);
  gemm_ffn1<<<(MT * 32 + 7) / 8, 256, 0, stream>>>(h2_h, w1_t, b1, ff1_h);
  gemm_ffn2<<<(MT * 8 + 7) / 8, 256, 0, stream>>>(ff1_h, w2_t, b2, x1, out);
}
